// SelfAttention_15960098472218
// MI455X (gfx1250) — matrix operand order repaired and checked by reading, not compiled, not run on hardware
//
#include <hip/hip_runtime.h>

#define BB 16
#define SS 512
#define DD 1024
#define HH 16
#define DHH 64
#define NUM_DEG 50
#define NUM_SPE 512

typedef __attribute__((ext_vector_type(16))) _Float16 v16h;
typedef __attribute__((ext_vector_type(8)))  _Float16 h8;
typedef __attribute__((ext_vector_type(8)))  float    v8f;

typedef int i4_t __attribute__((vector_size(16)));
typedef __attribute__((address_space(1))) i4_t gbl_i4_t;
typedef __attribute__((address_space(3))) i4_t lds_i4_t;

#if __has_builtin(__builtin_amdgcn_global_load_async_to_lds_b128) && \
    __has_builtin(__builtin_amdgcn_s_wait_asynccnt)
#define HAVE_ASYNC_LDS 1
#else
#define HAVE_ASYNC_LDS 0
#endif

// Copy 16 bytes global -> LDS via the CDNA5 async-DMA path (ASYNCcnt-tracked).
static __device__ __forceinline__ void cp16_g2s(void* dst_lds, const void* src_glob) {
#if HAVE_ASYNC_LDS
  __builtin_amdgcn_global_load_async_to_lds_b128(
      (gbl_i4_t*)src_glob, (lds_i4_t*)dst_lds, 0, 0);
#else
  *(h8*)dst_lds = *(const h8*)src_glob;
#endif
}
// Wait until at most N async ops remain outstanding (ASYNCcnt completes in
// order, so N = ops-in-flight for the *next* chunk gives double-buffer overlap).
template <int N> static __device__ __forceinline__ void cp_wait() {
#if HAVE_ASYNC_LDS
  __builtin_amdgcn_s_wait_asynccnt(N);
#endif
}

static __device__ __forceinline__ int lane_id() { return threadIdx.x & 31; }

// A-matrix (16x32 f16, M x K) loader from row-major LDS (stride ld halfs).
// Lane layout (ISA 7.12.2): m = lane%16, g = lane/16; elements 0..7 hold
// K = g*8..g*8+7, elements 8..15 hold K = 16+g*8..16+g*8+7 (two contiguous runs).
static __device__ __forceinline__ v16h load_a_lds(const _Float16* base, int ld) {
  const int l = lane_id();
  const int m = l & 15, g = l >> 4;
  h8 x0 = *(const h8*)(base + m * ld + g * 8);
  h8 x1 = *(const h8*)(base + m * ld + 16 + g * 8);
  return __builtin_shufflevector(x0, x1, 0,1,2,3,4,5,6,7,8,9,10,11,12,13,14,15);
}

// Same A layout, but source is f32 in LDS (probs tile) converted on the fly.
static __device__ __forceinline__ v16h load_a_f32(const float* base, int ld) {
  const int l = lane_id();
  const int m = l & 15, g = l >> 4;
  const float* p0 = base + m * ld + g * 8;
  const float* p1 = base + m * ld + 16 + g * 8;
  v16h r;
#pragma unroll
  for (int e = 0; e < 8; ++e) { r[e] = (_Float16)p0[e]; r[8 + e] = (_Float16)p1[e]; }
  return r;
}

// B-matrix (32x16 f16, K x N) loader from row-major LDS (stride ld halfs).
// Lane: n = lane%16, g = lane/16; elements 0..7 hold K = g*8..g*8+7,
// elements 8..15 hold K = 16+g*8..16+g*8+7 (same K order as A).
static __device__ __forceinline__ v16h load_b_lds(const _Float16* base, int ld) {
  const int l = lane_id();
  const int n = l & 15, g = l >> 4;
  v16h r;
#pragma unroll
  for (int e = 0; e < 8; ++e) {
    r[e] = base[(g * 8 + e) * ld + n];
    r[8 + e] = base[(16 + g * 8 + e) * ld + n];
  }
  return r;
}

static __device__ __forceinline__ v8f wmma_f16(v16h a, v16h b, v8f c) {
  return __builtin_amdgcn_wmma_f32_16x16x32_f16(false, a, false, b, (short)0, c, false, false);
}

// ---------------------------------------------------------------------------
// Kernel 1: h = hidden + (W_in*padmask)[in_deg] + (W_out*padmask)[out_deg] -> f16
// ---------------------------------------------------------------------------
__global__ __launch_bounds__(256) void prep_h_kernel(
    const float* __restrict__ hid, const int* __restrict__ indeg,
    const int* __restrict__ outdeg, const float* __restrict__ Win,
    const float* __restrict__ Wout, _Float16* __restrict__ h16) {
  const int row = blockIdx.x;               // b*S + s
  const int id = indeg[row], od = outdeg[row];
  const float mi = (id != 0) ? 1.f : 0.f;   // padding_idx = 0
  const float mo = (od != 0) ? 1.f : 0.f;
  const float* wi = Win + (size_t)id * DD;
  const float* wo = Wout + (size_t)od * DD;
  const size_t base = (size_t)row * DD;
  for (int d = threadIdx.x; d < DD; d += 256) {
    float v = hid[base + d] + mi * wi[d] + mo * wo[d];
    h16[base + d] = (_Float16)v;
  }
}

// ---------------------------------------------------------------------------
// Kernel 2: Wq|Wk|Wv f32 -> f16 (contiguous 3*D*D buffer)
// ---------------------------------------------------------------------------
__global__ __launch_bounds__(256) void cvt_w_kernel(
    const float* __restrict__ wq, const float* __restrict__ wk,
    const float* __restrict__ wv, _Float16* __restrict__ dst) {
  const size_t N = (size_t)DD * DD;
  size_t i = (size_t)blockIdx.x * 256 + threadIdx.x;
  const float* src = (i < N) ? wq : (i < 2 * N ? wk : wv);
  dst[i] = (_Float16)src[i % N];
}

// ---------------------------------------------------------------------------
// Kernel 3: QKV projection GEMM. z = blockIdx.z selects Q/K/V.
// C[8192x1024] = h16 @ W + b ; double-buffered async K-chunks of 64.
//   Q,V -> [B,H,S,DH] f16 ; K -> [B,H,DH,S] f16 (transposed for scores B-matrix)
// ---------------------------------------------------------------------------
__global__ __launch_bounds__(256) void qkv_gemm_kernel(
    const _Float16* __restrict__ h16, const _Float16* __restrict__ w16all,
    const float* __restrict__ bq, const float* __restrict__ bk,
    const float* __restrict__ bv, _Float16* __restrict__ q16,
    _Float16* __restrict__ kT16, _Float16* __restrict__ v16g) {
  __shared__ _Float16 As[2][64][72];   // 64 x 64 (+8 pad), double-buffered
  __shared__ _Float16 Bs[2][64][72];
  const int z = blockIdx.z;
  const int m0 = blockIdx.x * 64;
  const int n0 = blockIdx.y * 64;
  const _Float16* w16 = w16all + (size_t)z * DD * DD;
  const float* bsel = (z == 0) ? bq : (z == 1 ? bk : bv);
  const int t = threadIdx.x;
  const int wave = t >> 5;
  const int sr = t >> 2, sc4 = (t & 3) * 16;   // staging: 64 rows x 64 cols
  v8f acc0 = {}; v8f acc1 = {};

  auto stage = [&](int buf, int k0) {          // 4 async b128 ops per thread
    const _Float16* ga = &h16[(size_t)(m0 + sr) * DD + k0 + sc4];
    const _Float16* gb = &w16[(size_t)(k0 + sr) * DD + n0 + sc4];
    cp16_g2s(&As[buf][sr][sc4],     ga);
    cp16_g2s(&As[buf][sr][sc4 + 8], ga + 8);
    cp16_g2s(&Bs[buf][sr][sc4],     gb);
    cp16_g2s(&Bs[buf][sr][sc4 + 8], gb + 8);
  };

  stage(0, 0);
  for (int k0 = 0; k0 < DD; k0 += 64) {
    const int cur = (k0 >> 6) & 1;
    if (k0 + 64 < DD) { stage(cur ^ 1, k0 + 64); cp_wait<4>(); }
    else              { cp_wait<0>(); }
    __syncthreads();
#pragma unroll
    for (int i = 0; i < 2; ++i) {
      const int tile = wave * 2 + i;
      const int mt = tile >> 2, nt = tile & 3;
      v16h a0 = load_a_lds(&As[cur][mt * 16][0],  72);
      v16h a1 = load_a_lds(&As[cur][mt * 16][32], 72);
      v16h b0 = load_b_lds(&Bs[cur][0][nt * 16],  72);
      v16h b1 = load_b_lds(&Bs[cur][32][nt * 16], 72);
      v8f acc = i ? acc1 : acc0;
      acc = wmma_f16(a0, b0, acc);
      acc = wmma_f16(a1, b1, acc);
      if (i) acc1 = acc; else acc0 = acc;
    }
    __syncthreads();
  }

  const int l = lane_id();
  const int en = l & 15, em = (l >> 4) * 8;   // C layout: M = em + r, N = en
#pragma unroll
  for (int i = 0; i < 2; ++i) {
    const int tile = wave * 2 + i;
    const int mt = tile >> 2, nt = tile & 3;
    v8f acc = i ? acc1 : acc0;
#pragma unroll
    for (int r = 0; r < 8; ++r) {
      const int row = m0 + mt * 16 + em + r;       // b*S + s
      const int col = n0 + nt * 16 + en;           // h*DH + dh
      const _Float16 hv = (_Float16)(acc[r] + bsel[col]);
      const int bb = row >> 9, s = row & (SS - 1);
      const int hh = col >> 6, dh = col & (DHH - 1);
      const size_t bh = (size_t)bb * HH + hh;
      if (z == 0)      q16 [(bh * SS + s) * DHH + dh] = hv;
      else if (z == 1) kT16[(bh * DHH + dh) * SS + s] = hv;
      else             v16g[(bh * SS + s) * DHH + dh] = hv;
    }
  }
}

// ---------------------------------------------------------------------------
// Kernel 4: attention. One block = (b, h, 32-query tile). 8 waves.
//   scores (WMMA, f32 in LDS) -> softmax(+spe bias+mask, write probs) ->
//   ctx = P @ V (WMMA, probs converted f32->f16 on load)
// All operand staging is async-DMA, double-buffered.
// ---------------------------------------------------------------------------
__global__ __launch_bounds__(256) void attn_kernel(
    const _Float16* __restrict__ q16, const _Float16* __restrict__ kT16,
    const _Float16* __restrict__ v16g, const int* __restrict__ spe,
    const float* __restrict__ W_spe, const float* __restrict__ amask,
    float* __restrict__ ctx_out, float* __restrict__ probs_out) {
  __shared__ float    sc[32][512];     // score/prob tile (f32)
  __shared__ _Float16 qs[32][72];      // Q tile [32 x 64]
  __shared__ _Float16 ks[2][64][72];   // K^T chunks [64 dh x 64 keys], dbl-buf
  __shared__ _Float16 vs[2][32][72];   // V chunks [32 keys x 64 dh], dbl-buf
  __shared__ float    wspe_h[NUM_SPE]; // W_spe[:, h]
  __shared__ float    maskv[SS];

  const int m0 = blockIdx.x * 32;
  const int h  = blockIdx.y;
  const int b  = blockIdx.z;
  const size_t bh = (size_t)b * HH + h;
  const int t = threadIdx.x, wave = t >> 5, l = t & 31;
  const int mt = wave >> 2, nt = wave & 3;   // this wave's 16x16 tile in 32x64

  for (int v = t; v < NUM_SPE; v += 256) wspe_h[v] = W_spe[(size_t)v * HH + h];
  for (int j = t; j < SS; j += 256) maskv[j] = amask[(size_t)b * SS + j];
  { // stage Q tile (one-time)
    const int i = t >> 3, c = (t & 7) * 8;
    cp16_g2s(&qs[i][c], &q16[(bh * SS + (m0 + i)) * DHH + c]);
  }
  cp_wait<0>();
  __syncthreads();

  // ---------------- scores = Q K^T ----------------
  auto stage_k = [&](int buf, int n0c) {       // 2 async ops per thread
    const int dh = t >> 2, c = (t & 3) * 16;
    const _Float16* src = &kT16[(bh * DHH + dh) * SS + n0c + c];
    cp16_g2s(&ks[buf][dh][c],     src);
    cp16_g2s(&ks[buf][dh][c + 8], src + 8);
  };
  stage_k(0, 0);
  for (int n0c = 0; n0c < SS; n0c += 64) {
    const int cur = (n0c >> 6) & 1;
    if (n0c + 64 < SS) { stage_k(cur ^ 1, n0c + 64); cp_wait<2>(); }
    else               { cp_wait<0>(); }
    __syncthreads();
    v8f c8 = {};
    v16h a0 = load_a_lds(&qs[mt * 16][0], 72);
    v16h b0 = load_b_lds(&ks[cur][0][nt * 16], 72);
    c8 = wmma_f16(a0, b0, c8);
    v16h a1 = load_a_lds(&qs[mt * 16][32], 72);
    v16h b1 = load_b_lds(&ks[cur][32][nt * 16], 72);
    c8 = wmma_f16(a1, b1, c8);
    const int en = l & 15, em = (l >> 4) * 8;
#pragma unroll
    for (int r = 0; r < 8; ++r)
      sc[mt * 16 + em + r][n0c + nt * 16 + en] = c8[r];
    __syncthreads();
  }

  // ---------------- softmax (8 lanes per row) ----------------
  {
    const int row = wave * 4 + (l >> 3);         // 0..31
    const int p = l & 7;                         // 64-wide chunk per lane
    const int s_glob = m0 + row;
    const int* spe_row = spe + ((size_t)b * SS + s_glob) * SS;
    float* scr = sc[row];
    const float scale = 0.125f;                  // 1/sqrt(64)
    float mx = -3.4e38f;
    for (int j = p * 64; j < p * 64 + 64; ++j) {
      float logit = scr[j] * scale + wspe_h[spe_row[j]] + maskv[j];
      scr[j] = logit;
      mx = fmaxf(mx, logit);
    }
#pragma unroll
    for (int d = 1; d < 8; d <<= 1) mx = fmaxf(mx, __shfl_xor(mx, d, 32));
    float sum = 0.f;
    for (int j = p * 64; j < p * 64 + 64; ++j) {
      float e = __expf(scr[j] - mx);
      scr[j] = e;
      sum += e;
    }
#pragma unroll
    for (int d = 1; d < 8; d <<= 1) sum += __shfl_xor(sum, d, 32);
    const float inv = 1.f / sum;
    float* prow = probs_out + (bh * SS + s_glob) * SS;
    for (int j = p * 64; j < p * 64 + 64; ++j) {
      float pv = scr[j] * inv;
      scr[j] = pv;
      prow[j] = pv;
    }
  }
  __syncthreads();

  // ---------------- ctx = P @ V ----------------
  auto stage_v = [&](int buf, int k0c) {       // 1 async op per thread
    const int r = t >> 3, c = (t & 7) * 8;
    cp16_g2s(&vs[buf][r][c], &v16g[(bh * SS + k0c + r) * DHH + c]);
  };
  v8f cacc = {};
  stage_v(0, 0);
  for (int k0c = 0; k0c < SS; k0c += 32) {
    const int cur = (k0c >> 5) & 1;
    if (k0c + 32 < SS) { stage_v(cur ^ 1, k0c + 32); cp_wait<1>(); }
    else               { cp_wait<0>(); }
    __syncthreads();
    v16h a = load_a_f32(&sc[mt * 16][k0c], 512);
    v16h bm = load_b_lds(&vs[cur][0][nt * 16], 72);
    cacc = wmma_f16(a, bm, cacc);
    __syncthreads();
  }
  {
    const int en = l & 15, em = (l >> 4) * 8;
#pragma unroll
    for (int r = 0; r < 8; ++r) {
      const int s = m0 + mt * 16 + em + r;
      const int dcol = h * DHH + nt * 16 + en;
      ctx_out[((size_t)b * SS + s) * DD + dcol] = cacc[r];
    }
  }
}

// ---------------------------------------------------------------------------
extern "C" void kernel_launch(void* const* d_in, const int* in_sizes, int n_in,
                              void* d_out, int out_size, void* d_ws, size_t ws_size,
                              hipStream_t stream) {
  (void)in_sizes; (void)n_in; (void)out_size; (void)ws_size;
  const float* hidden = (const float*)d_in[0];
  const float* amask  = (const float*)d_in[1];
  const int*   indeg  = (const int*)d_in[3];
  const int*   outdeg = (const int*)d_in[4];
  const int*   spe    = (const int*)d_in[5];
  const float* Wq  = (const float*)d_in[8];
  const float* bq  = (const float*)d_in[9];
  const float* Wk  = (const float*)d_in[10];
  const float* bk  = (const float*)d_in[11];
  const float* Wv  = (const float*)d_in[12];
  const float* bv  = (const float*)d_in[13];
  const float* Win = (const float*)d_in[14];
  const float* Wou = (const float*)d_in[15];
  const float* Wsp = (const float*)d_in[16];

  char* ws = (char*)d_ws;
  _Float16* h16  = (_Float16*)(ws);                 // 16.78 MB
  _Float16* w16  = (_Float16*)(ws + 16777216);      //  6.29 MB
  _Float16* q16  = (_Float16*)(ws + 23068672);      // 16.78 MB
  _Float16* kT16 = (_Float16*)(ws + 39845888);      // 16.78 MB
  _Float16* v16g = (_Float16*)(ws + 56623104);      // 16.78 MB (total 73.4 MB)

  float* ctx   = (float*)d_out;                         // [B,S,D]
  float* probs = ctx + (size_t)BB * SS * DD;            // [B,H,S,S]

  prep_h_kernel<<<BB * SS, 256, 0, stream>>>(hidden, indeg, outdeg, Win, Wou, h16);
  cvt_w_kernel<<<(3 * DD * DD) / 256, 256, 0, stream>>>(Wq, Wk, Wv, w16);
  qkv_gemm_kernel<<<dim3((BB * SS) / 64, DD / 64, 3), 256, 0, stream>>>(
      h16, w16, bq, bk, bv, q16, kT16, v16g);
  attn_kernel<<<dim3(SS / 32, HH, BB), 256, 0, stream>>>(
      q16, kT16, v16g, spe, Wsp, amask, ctx, probs);
}
